// lTFWG_82532091560415
// MI455X (gfx1250) — matrix operand order repaired and checked by reading, not compiled, not run on hardware
//
#include <hip/hip_runtime.h>
#include <hip/hip_bf16.h>
#include <math.h>

// ---------------------------------------------------------------------------
// FGW distances on MI455X (gfx1250, wave32, WMMA).
// One 256-thread block (8 wave32) per node; one wave per 16-column tile of the
// TN-padded (5->8) template axis, i.e. 2 templates per wave.
// ---------------------------------------------------------------------------

typedef __attribute__((ext_vector_type(16))) __bf16 v16bf;
typedef __attribute__((ext_vector_type(8)))  float  v8f;

#define NNODES 8192
#define NF     128
#define NK     16
#define NZ     16
#define NT     5
#define NTP    8      // padded TN
#define NEG_BIG (-1e30f)

__global__ __launch_bounds__(256, 2) void fgw_wmma_kernel(
    const float* __restrict__ x,      // (N, F)
    const float* __restrict__ Cm,     // (N, N)
    const float* __restrict__ tmpl,   // (Z, TN, TN)
    const float* __restrict__ tf,     // (Z, TN, F)
    const int*   __restrict__ nidx,   // (N, K)
    float*       __restrict__ out)    // (N, Z)
{
  __shared__ int   s_ni[NK];
  __shared__ float s_x[NK][NF];        // gathered x_loc (f32)
  __shared__ float s_C[NK][NK];        // gathered C_loc
  __shared__ float s_xn[NK];           // |x_loc[k]|^2
  __shared__ float s_hc1[NK];          // mean_j C^2
  __shared__ float s_hc2[NZ*NTP];      // mean_s tmpl^2 (padded cols = 0)
  __shared__ float s_tmpl[NZ*NT*NT];

  const int n   = blockIdx.x;
  const int tid = threadIdx.x;

  // ---- stage neighbor indices, zero accumulators ----
  if (tid < NK) {
    s_ni[tid]  = nidx[n*NK + tid];
    s_xn[tid]  = 0.f;
    s_hc1[tid] = 0.f;
  }
  __syncthreads();

  // ---- cooperative gather: x_loc, C_loc, norms, templates ----
  {
    const int k = tid >> 4;        // neighbor row 0..15
    const int c = tid & 15;        // 8-float chunk 0..15
    const float* xr = x + (size_t)s_ni[k]*NF + c*8;
    float4 a0 = ((const float4*)xr)[0];
    float4 a1 = ((const float4*)xr)[1];
    ((float4*)&s_x[k][c*8])[0] = a0;
    ((float4*)&s_x[k][c*8])[1] = a1;
    float sq = a0.x*a0.x + a0.y*a0.y + a0.z*a0.z + a0.w*a0.w
             + a1.x*a1.x + a1.y*a1.y + a1.z*a1.z + a1.w*a1.w;
    atomicAdd(&s_xn[k], sq);

    float cv = Cm[(size_t)s_ni[k]*NNODES + s_ni[c]];
    s_C[k][c] = cv;
    atomicAdd(&s_hc1[k], cv*cv*(1.f/(float)NK));

    if (tid < NZ*NT*NT) s_tmpl[tid] = tmpl[tid];
    if (tid < NZ*NTP) {
      int zz = tid >> 3, tt = tid & 7;
      float h = 0.f;
      if (tt < NT) {
        for (int s = 0; s < NT; ++s) {
          float w = tmpl[(zz*NT + s)*NT + tt];
          h += w*w;
        }
        h *= (1.f/(float)NT);
      }
      s_hc2[tid] = h;
    }
  }
  __syncthreads();

  // ---- per-wave setup: wave ct owns padded columns [16*ct, 16*ct+16) ----
  const int lane = tid & 31;
  const int ct   = tid >> 5;         // tile 0..7 (2 templates each)
  const int col  = lane & 15;        // D-layout column
  const int half = lane >> 4;        // D-layout row block (0: M 0..7, 1: M 8..15)
  const int zt   = ct*16 + col;      // padded (z,t) column index
  const int z    = zt >> 3;
  const int tcol = zt & 7;           // t within template (>=5 => padding)
  const bool valid = (tcol < NT);

  // ---- M tile: D = x_loc(16x128) * tf^T(128x16) via 4 chained bf16 WMMAs ----
  v8f acc = {};
  float tfn = 0.f;
  const float* tfrow = tf + ((size_t)z*NT + (valid ? tcol : 0))*NF;
  #pragma unroll
  for (int kc = 0; kc < 4; ++kc) {
    v16bf afrag, bfrag;
    // A layout (16-bit, 16x32): lane<16 K = {e, 16+e-8}; lane>=16 K = {8+e, 24+e-8}
    #pragma unroll
    for (int e = 0; e < 8; ++e) {
      afrag[e]   = (__bf16)s_x[col][kc*32 +      half*8 + e];
      afrag[e+8] = (__bf16)s_x[col][kc*32 + 16 + half*8 + e];
    }
    // B layout (16-bit, 32x16): lane<16 col=lane, K = {e, 16+e-8}; lane>=16 col=lane-16, K = {8+e, 24+e-8}
    if (valid) {
      #pragma unroll
      for (int e = 0; e < 16; ++e) {
        float f = tfrow[kc*32 + (e < 8 ? half*8 + e : 16 + half*8 + (e - 8))];
        bfrag[e] = (__bf16)f;
        tfn += f*f;
      }
    } else {
      #pragma unroll
      for (int e = 0; e < 16; ++e) bfrag[e] = (__bf16)0.f;
    }
    acc = __builtin_amdgcn_wmma_f32_16x16x32_bf16(
        false, afrag, false, bfrag, (short)0, acc, false, false);
  }
  tfn += __shfl_xor(tfn, 16, 32);    // lane halves covered disjoint feature ranges

  // ---- assemble M, constC pieces in D-layout registers ----
  float mm[8], hc1v[8];
  const float hc2 = s_hc2[zt];
  #pragma unroll
  for (int v = 0; v < 8; ++v) {
    int kr = half*8 + v;             // D row = neighbor k
    hc1v[v] = s_hc1[kr];
    mm[v]   = s_xn[kr] + tfn - 2.f*acc[v];
  }

  // ---- C_loc as bf16 A-fragment (K=j, 16 real + 16 zero-pad), exact in bf16 ----
  v16bf cfrag;
  #pragma unroll
  for (int e = 0; e < 8; ++e) {
    cfrag[e]   = (__bf16)s_C[col][half*8 + e];
    cfrag[e+8] = (__bf16)0.f;
  }

  // templates weights for the (5x5) contraction: tw[j] = tmpl[z][s=tcol][j]
  float tw[NT];
  #pragma unroll
  for (int j = 0; j < NT; ++j)
    tw[j] = valid ? s_tmpl[(z*NT + tcol)*NT + j] : 0.f;

  float T[8];
  #pragma unroll
  for (int v = 0; v < 8; ++v) T[v] = valid ? (1.f/(float)(NK*NT)) : 0.f;

  const float LOGP = -2.77258872f;   // -log K
  const float LOGQ = -1.60943791f;   // -log TN
  const int   gb   = lane & ~7;      // base lane of this template's 8-lane group

  float Ac[8];
  float lk[8];
  float u[8];
  float vc = 0.f;

  #pragma unroll 1
  for (int it = 0; it < 5; ++it) {
    // ---- cross(T): W = C_loc x T (one WMMA), then A = W x tmpl^T (bpermute) ----
    v16bf tb;
    #pragma unroll
    for (int e = 0; e < 8; ++e) {
      tb[e] = (__bf16)T[e]; tb[e+8] = (__bf16)0.f; // K=16..31 pad
    }
    v8f W = {};
    W = __builtin_amdgcn_wmma_f32_16x16x32_bf16(
        false, cfrag, false, tb, (short)0, W, false, false);
    #pragma unroll
    for (int v = 0; v < 8; ++v) {
      float s = 0.f;
      #pragma unroll
      for (int j = 0; j < NT; ++j)
        s += __shfl(W[v], gb + j, 32) * tw[j];   // ds_bpermute within 8-lane group
      Ac[v] = s;
    }
    if (it == 4) break;   // final cross only

    // ---- grad -> logK (ALPHA=0.5, EPS=1):  lk = -(0.5*M + (constC - 2A)) ----
    #pragma unroll
    for (int v = 0; v < 8; ++v) {
      float g = 0.5f*mm[v] + (hc1v[v] + hc2 - 2.f*Ac[v]);
      lk[v] = valid ? -g : NEG_BIG;
    }

    // ---- log-domain Sinkhorn, fully in registers ----
    #pragma unroll
    for (int v = 0; v < 8; ++v) u[v] = 0.f;
    vc = 0.f;
    #pragma unroll 1
    for (int sk = 0; sk < 5; ++sk) {
      // v-update: LSE over k = 8 VGPR rows + half swap
      float m1 = lk[0] + u[0];
      #pragma unroll
      for (int v = 1; v < 8; ++v) m1 = fmaxf(m1, lk[v] + u[v]);
      float m2 = fmaxf(m1, __shfl_xor(m1, 16, 32));
      float ss = 0.f;
      #pragma unroll
      for (int v = 0; v < 8; ++v) ss += __expf(lk[v] + u[v] - m2);
      ss += __shfl_xor(ss, 16, 32);
      vc = LOGQ - (m2 + __logf(ss));
      if (!valid) vc = NEG_BIG;      // padded columns contribute exp(-inf)=0
      // u-update: LSE over t = butterfly across the 8-lane group
      #pragma unroll
      for (int v = 0; v < 8; ++v) {
        float val = lk[v] + vc;
        float m8 = val;
        m8 = fmaxf(m8, __shfl_xor(m8, 1, 32));
        m8 = fmaxf(m8, __shfl_xor(m8, 2, 32));
        m8 = fmaxf(m8, __shfl_xor(m8, 4, 32));
        float e8 = __expf(val - m8);
        e8 += __shfl_xor(e8, 1, 32);
        e8 += __shfl_xor(e8, 2, 32);
        e8 += __shfl_xor(e8, 4, 32);
        u[v] = LOGP - (m8 + __logf(e8));
      }
    }

    // ---- Frank-Wolfe update ----
    const float gamma = 2.f/(float)(it + 2);
    #pragma unroll
    for (int v = 0; v < 8; ++v) {
      float tl = __expf(lk[v] + u[v] + vc);   // 0 on padded columns
      T[v] = (1.f - gamma)*T[v] + gamma*tl;
    }
  }

  // ---- objective: sum T * (0.5*M + 0.5*(constC - 2A)) over (k, t) ----
  float part = 0.f;
  #pragma unroll
  for (int v = 0; v < 8; ++v)
    part += T[v] * (0.5f*mm[v] + 0.5f*(hc1v[v] + hc2 - 2.f*Ac[v]));
  part += __shfl_xor(part, 1, 32);
  part += __shfl_xor(part, 2, 32);
  part += __shfl_xor(part, 4, 32);
  part += __shfl_xor(part, 16, 32);

  if (half == 0 && (lane & 7) == 0)
    out[(size_t)n*NZ + ct*2 + (lane >> 3)] = part;
}

extern "C" void kernel_launch(void* const* d_in, const int* in_sizes, int n_in,
                              void* d_out, int out_size, void* d_ws, size_t ws_size,
                              hipStream_t stream) {
  (void)in_sizes; (void)n_in; (void)out_size; (void)d_ws; (void)ws_size;
  const float* x    = (const float*)d_in[0];
  const float* Cm   = (const float*)d_in[1];
  const float* tmpl = (const float*)d_in[2];
  const float* tf   = (const float*)d_in[3];
  const int*   ni   = (const int*)  d_in[4];
  float*       out  = (float*)d_out;
  fgw_wmma_kernel<<<NNODES, 256, 0, stream>>>(x, Cm, tmpl, tf, ni, out);
}
